// DecoderOnlyTransformer_10840497455507
// MI455X (gfx1250) — matrix operand order repaired and checked by reading, not compiled, not run on hardware
//
#include <hip/hip_runtime.h>
#include <hip/hip_bf16.h>
#include <stdint.h>

// ---------------------------------------------------------------------------
// DecoderOnlyTransformer forward for MI455X (gfx1250, wave32, WMMA).
// - all GEMMs: v_wmma_f32_16x16x32_bf16 (f32 accumulation)
// - B tiles staged in LDS via async global->LDS copies (ASYNCcnt), double
//   buffered so the copy of stage k+1 overlaps the WMMAs of stage k
// - logits (512 MB f32) written with non-temporal stores so the 131 MB wfc
//   stays resident in the 192 MB L2
// ---------------------------------------------------------------------------

typedef __attribute__((ext_vector_type(16))) __bf16 v16bf;
typedef __attribute__((ext_vector_type(8)))  __bf16 v8bf;
typedef __attribute__((ext_vector_type(8)))  float  v8f;
typedef __attribute__((ext_vector_type(4)))  int    v4i;

#define WMMA_BF16(a, b, c) \
  __builtin_amdgcn_wmma_f32_16x16x32_bf16(false, (a), false, (b), (short)0, (c), false, false)

// Async 16-byte global -> LDS copy (gfx1250 GLOBAL_LOAD_ASYNC_TO_LDS_B128).
// Builtin signature (from clang diagnostic): (v4i __device__*, v4i __shared__*,
// imm offset, imm cpol) -- i.e. (global src, LDS dst, ...).
__device__ __forceinline__ void async_copy_b128(const float* g, float* l) {
#if __has_builtin(__builtin_amdgcn_global_load_async_to_lds_b128)
  __builtin_amdgcn_global_load_async_to_lds_b128(
      (__attribute__((address_space(1))) v4i*)(void*)(g),
      (__attribute__((address_space(3))) v4i*)(void*)(l), 0, 0);
#else
  asm volatile("global_load_async_to_lds_b128 %0, %1, off"
               :: "v"((unsigned)(uintptr_t)l),
                  "v"((unsigned long long)(uintptr_t)g)
               : "memory");
#endif
}

__device__ __forceinline__ void wait_async0() {
#if __has_builtin(__builtin_amdgcn_s_wait_asynccnt)
  __builtin_amdgcn_s_wait_asynccnt(0);
#else
  asm volatile("s_wait_asynccnt 0x0" ::: "memory");
#endif
}

// ---- Operand loaders, contiguous k order (k = 16*(lane/16) + s) in A and B alike --------
// A-matrix 16x32 bf16: lane m(0..15) holds K 0..15; lane m+16 holds
// K 16..31.  Row-major source, ld in elements.
__device__ __forceinline__ v16bf load_A_bf16(const __bf16* A, int ld, int lane) {
  const int m  = lane & 15;
  const int hi = lane >> 4;
  const __bf16* p = A + (size_t)m * ld + hi * 16;
  v8bf lo = *(const v8bf*)(p);
  v8bf hh = *(const v8bf*)(p + 8);
  return __builtin_shufflevector(lo, hh, 0,1,2,3,4,5,6,7,8,9,10,11,12,13,14,15);
}

// B-matrix 32x16 bf16: lane n(0..15) = column n, K 0..15; lane n+16: K 16..31.
// Source is "column-major for B" == row-major [N,K] (our Kmat / V^T rows).
__device__ __forceinline__ v16bf load_B_bf16(const __bf16* B, int ld, int lane) {
  const int n  = lane & 15;
  const int hi = lane >> 4;
  const __bf16* p = B + (size_t)n * ld + hi * 16;
  v8bf a = *(const v8bf*)(p);
  v8bf b = *(const v8bf*)(p + 8);
  return __builtin_shufflevector(a, b, 0,1,2,3,4,5,6,7,8,9,10,11,12,13,14,15);
}

// Same B layout, f32 source (raw weights in LDS or global): convert in-flight.
__device__ __forceinline__ v16bf load_B_f32(const float* B, int ld, int lane) {
  const int n  = lane & 15;
  const int hi = lane >> 4;
  const float4* p = (const float4*)(B + (size_t)n * ld + hi * 16);
  v16bf r;
#pragma unroll
  for (int i = 0; i < 4; ++i) {
    float4 f = p[i];
    r[4 * i + 0] = (__bf16)f.x;
    r[4 * i + 1] = (__bf16)f.y;
    r[4 * i + 2] = (__bf16)f.z;
    r[4 * i + 3] = (__bf16)f.w;
  }
  return r;
}

// ---------------------------------------------------------------------------
// Kernel 1: x = emb[idx]  (f32 copy for residual + bf16 copy for GEMM A)
// ---------------------------------------------------------------------------
__global__ void embed_gather(const int* __restrict__ idx,
                             const float* __restrict__ emb,
                             float* __restrict__ xf, __bf16* __restrict__ xb) {
  const int n   = blockIdx.x;            // token 0..4095
  const int row = idx[n];
  const float* src = emb + (size_t)row * 1024;
  for (int e = threadIdx.x; e < 1024; e += blockDim.x) {
    float v = src[e];
    xf[(size_t)n * 1024 + e] = v;
    xb[(size_t)n * 1024 + e] = (__bf16)v;
  }
}

// ---------------------------------------------------------------------------
// Kernel 2: C[M,N] = A[M,K](bf16) @ B[N,K]^T(f32)
// Block = 8 waves sharing one 64-row B tile (waves = 8 consecutive M-tiles).
// B tile (64x32 f32 = 8KB) staged in LDS by async copies, double buffered:
//   wait(stage k) -> barrier -> issue(stage k+1) -> WMMAs on stage k.
// Requires M % 128 == 0, N % 64 == 0, K % 32 == 0 (true for all uses).
// EPI 0: store bf16   EPI 1: store bf16, += Res(f32)   EPI 2: store f32 (NT)
// ---------------------------------------------------------------------------
template <int EPI>
__global__ __launch_bounds__(256) void gemm_bf16_nt(
    const __bf16* __restrict__ A, const float* __restrict__ B,
    void* __restrict__ Cout, const float* __restrict__ Res,
    int M, int N, int K) {
  __shared__ __align__(16) float sB[2][64 * 32];   // 2 x 8 KB

  const int tid  = threadIdx.x;
  const int wv   = tid >> 5;
  const int lane = tid & 31;
  const int n64  = N >> 6;
  const int nt   = blockIdx.x % n64;               // shared N-tile (64 cols)
  const int mt   = (blockIdx.x / n64) * 8 + wv;    // per-wave M-tile (16 rows)

  const float* gB0 = B + (size_t)(nt * 64) * K;

  // Stage one 64x32 f32 B tile into sB[buf]: 512 x 16B transfers, 2/thread.
  auto stage = [&](int buf, int k0) {
#pragma unroll
    for (int it = 0; it < 2; ++it) {
      const int j   = tid + it * 256;   // 0..511
      const int row = j >> 3;           // 0..63
      const int seg = j & 7;            // 8 x 4 floats per row
      async_copy_b128(gB0 + (size_t)row * K + k0 + seg * 4,
                      &sB[buf][row * 32 + seg * 4]);
    }
  };

  v8f acc[4] = {};
  const __bf16* Arow = A + (size_t)(mt * 16) * K;

  stage(0, 0);
  int buf = 0;
  for (int k0 = 0; k0 < K; k0 += 32) {
    wait_async0();          // our wave's async copies for this stage done
    __syncthreads();        // everyone's copies visible / last reads retired
    if (k0 + 32 < K) stage(buf ^ 1, k0 + 32);   // overlap next copy w/ math

    v16bf a = load_A_bf16(Arow + k0, K, lane);
#pragma unroll
    for (int t = 0; t < 4; ++t) {
      v16bf b = load_B_f32(&sB[buf][(t * 16) * 32], 32, lane);
      acc[t] = WMMA_BF16(a, b, acc[t]);
    }
    buf ^= 1;
  }

  const int hi = lane >> 4, ln = lane & 15;
#pragma unroll
  for (int t = 0; t < 4; ++t) {
#pragma unroll
    for (int j = 0; j < 8; ++j) {
      const int r = mt * 16 + j + 8 * hi;
      const int c = nt * 64 + t * 16 + ln;
      float v = acc[t][j];
      if (EPI == 1) v += Res[(size_t)r * N + c];
      if (EPI == 2) {
        // 512 MB logits stream: NT stores keep wfc resident in L2
        __builtin_nontemporal_store(v, &((float*)Cout)[(size_t)r * N + c]);
      } else {
        ((__bf16*)Cout)[(size_t)r * N + c] = (__bf16)v;
      }
    }
  }
}

// ---------------------------------------------------------------------------
// Kernel 3: V^T for the PV matmul (so B-operand loads are contiguous)
// V: [B*T, E] (col = h*64+d)  ->  VT: [B*H, D, T]
// ---------------------------------------------------------------------------
__global__ void transpose_v(const __bf16* __restrict__ V, __bf16* __restrict__ VT) {
  const int i = blockIdx.x * blockDim.x + threadIdx.x;   // over 2*16*64*2048
  if (i >= 2 * 16 * 64 * 2048) return;
  const int t  = i & 2047;
  const int d  = (i >> 11) & 63;
  const int bh = i >> 17;
  const int b = bh >> 4, h = bh & 15;
  VT[i] = V[((size_t)(b * 2048 + t)) * 1024 + h * 64 + d];
}

// ---------------------------------------------------------------------------
// Kernel 4: flash attention.  One wave per (b,h, 16-row Q tile).
// S = QK^T/8 (WMMA) -> online softmax (shfl_xor in 16-lane halves) ->
// P C-layout -> LDS -> A-layout -> O += P@V (WMMA vs V^T).
// ---------------------------------------------------------------------------
__global__ __launch_bounds__(32) void flash_attn(
    const __bf16* __restrict__ Q, const __bf16* __restrict__ Km,
    const __bf16* __restrict__ VT, __bf16* __restrict__ O) {
  __shared__ __align__(16) __bf16 sP[16][32];

  const int lane = threadIdx.x & 31;
  const int mt = blockIdx.x;            // 0..127 (T/16)
  const int bh = blockIdx.y;            // 0..31
  const int b = bh >> 4, h = bh & 15;
  const int T = 2048, E = 1024, D = 64;
  const int rbase = mt * 16;

  const __bf16* qp  = Q  + ((size_t)(b * T + rbase)) * E + h * D;
  const __bf16* kp  = Km + ((size_t)(b * T)) * E + h * D;
  const __bf16* vtp = VT + ((size_t)bh * D) * T;

  // Q tile (16x64) as two 16x32 A operands, reused for every S tile.
  const v16bf aq0 = load_A_bf16(qp + 0,  E, lane);
  const v16bf aq1 = load_A_bf16(qp + 32, E, lane);

  const int hi = lane >> 4, ln = lane & 15;
  float mrow[8], lrow[8];
  v8f Oacc[4] = {};
#pragma unroll
  for (int j = 0; j < 8; ++j) { mrow[j] = -1e30f; lrow[j] = 0.0f; }

  const int nst = (rbase + 15) / 32 + 1;      // causal: cover s <= rbase+15
  for (int st = 0; st < nst; ++st) {
    const int s0 = st * 32;
    // ---- S = Q K^T over d=64 (two K-steps), 32 columns = two C blocks ----
    v8f sa = {}, sb = {};
    {
      v16bf b00 = load_B_bf16(kp + (size_t)s0 * E + 0,         E, lane);
      v16bf b10 = load_B_bf16(kp + (size_t)(s0 + 16) * E + 0,  E, lane);
      sa = WMMA_BF16(aq0, b00, sa);
      sb = WMMA_BF16(aq0, b10, sb);
      v16bf b01 = load_B_bf16(kp + (size_t)s0 * E + 32,        E, lane);
      v16bf b11 = load_B_bf16(kp + (size_t)(s0 + 16) * E + 32, E, lane);
      sa = WMMA_BF16(aq1, b01, sa);
      sb = WMMA_BF16(aq1, b11, sb);
    }
    // ---- scale, causal mask, online softmax (per row j+8*hi) ----
    float pr0[8], pr1[8];
#pragma unroll
    for (int j = 0; j < 8; ++j) {
      const int r  = rbase + j + 8 * hi;
      const int c0 = s0 + ln, c1 = c0 + 16;
      float x0 = sa[j] * 0.125f;              // 1/sqrt(64)
      float x1 = sb[j] * 0.125f;
      if (c0 > r) x0 = -1e30f;
      if (c1 > r) x1 = -1e30f;
      float mx = fmaxf(x0, x1);
      mx = fmaxf(mx, __shfl_xor(mx, 1));
      mx = fmaxf(mx, __shfl_xor(mx, 2));
      mx = fmaxf(mx, __shfl_xor(mx, 4));
      mx = fmaxf(mx, __shfl_xor(mx, 8));      // 16-lane half reduction (wave32)
      const float mnew  = fmaxf(mrow[j], mx);
      const float alpha = __expf(mrow[j] - mnew);
      const float p0 = __expf(x0 - mnew);
      const float p1 = __expf(x1 - mnew);
      float rs = p0 + p1;
      rs += __shfl_xor(rs, 1);
      rs += __shfl_xor(rs, 2);
      rs += __shfl_xor(rs, 4);
      rs += __shfl_xor(rs, 8);
      lrow[j] = lrow[j] * alpha + rs;
      mrow[j] = mnew;
      pr0[j] = p0; pr1[j] = p1;
#pragma unroll
      for (int t = 0; t < 4; ++t) Oacc[t][j] = Oacc[t][j] * alpha;
    }
    // ---- transpose P (C-layout -> A-layout) through LDS ----
#pragma unroll
    for (int j = 0; j < 8; ++j) {
      sP[j + 8 * hi][ln]      = (__bf16)pr0[j];
      sP[j + 8 * hi][ln + 16] = (__bf16)pr1[j];
    }
    __syncthreads();
    const v16bf ap = load_A_bf16(&sP[0][0], 32, lane);
    // ---- O += P @ V  (B operand from V^T rows: contiguous in s) ----
#pragma unroll
    for (int t = 0; t < 4; ++t) {
      v16bf bv = load_B_bf16(vtp + (size_t)(t * 16) * T + s0, T, lane);
      Oacc[t] = WMMA_BF16(ap, bv, Oacc[t]);
    }
    __syncthreads();
  }

  // ---- normalize and store O in [4096, 1024] bf16 (row = b*T+t, col = h*64+d)
#pragma unroll
  for (int j = 0; j < 8; ++j) {
    const float inv = 1.0f / lrow[j];
    const size_t r = (size_t)(b * T + rbase + j + 8 * hi);
#pragma unroll
    for (int t = 0; t < 4; ++t) {
      const int c = h * D + t * 16 + ln;
      O[r * E + c] = (__bf16)(Oacc[t][j] * inv);
    }
  }
}

// ---------------------------------------------------------------------------
// Host-side orchestration (graph-capture safe: only kernel launches on stream)
// ---------------------------------------------------------------------------
extern "C" void kernel_launch(void* const* d_in, const int* in_sizes, int n_in,
                              void* d_out, int out_size, void* d_ws, size_t ws_size,
                              hipStream_t stream) {
  (void)in_sizes; (void)n_in; (void)out_size; (void)ws_size;
  const int*   idx = (const int*)d_in[0];
  const float* emb = (const float*)d_in[1];
  const float* wq  = (const float*)d_in[2];
  const float* wk  = (const float*)d_in[3];
  const float* wv  = (const float*)d_in[4];
  const float* wo  = (const float*)d_in[5];
  const float* wfc = (const float*)d_in[6];
  float* logits = (float*)d_out;

  const int NT = 4096;   // B*T tokens
  const int E  = 1024;
  const int V  = 32000;

  char* ws = (char*)d_ws;
  size_t off = 0;
  auto carve = [&](size_t bytes) -> void* {
    off = (off + 255) & ~(size_t)255;
    void* p = ws + off;
    off += bytes;
    return p;
  };
  float*  xf   = (float*)carve((size_t)NT * E * 4);   // x (f32, residual)
  __bf16* xb   = (__bf16*)carve((size_t)NT * E * 2);  // x (bf16, GEMM A)
  __bf16* q    = (__bf16*)carve((size_t)NT * E * 2);
  __bf16* k    = (__bf16*)carve((size_t)NT * E * 2);
  __bf16* v    = (__bf16*)carve((size_t)NT * E * 2);
  __bf16* vt   = (__bf16*)carve((size_t)NT * E * 2);  // [B*H, 64, 2048]
  __bf16* attn = (__bf16*)carve((size_t)NT * E * 2);
  __bf16* xres = (__bf16*)carve((size_t)NT * E * 2);

  // 1) embedding gather
  embed_gather<<<NT, 256, 0, stream>>>(idx, emb, xf, xb);

  // 2) Q,K,V projections: blocks = (M/128) x (N/64) = 32*16 = 512
  const int projBlocks = (NT / 128) * (E / 64);
  gemm_bf16_nt<0><<<projBlocks, 256, 0, stream>>>(xb, wq, q, nullptr, NT, E, E);
  gemm_bf16_nt<0><<<projBlocks, 256, 0, stream>>>(xb, wk, k, nullptr, NT, E, E);
  gemm_bf16_nt<0><<<projBlocks, 256, 0, stream>>>(xb, wv, v, nullptr, NT, E, E);

  // 3) V transpose for contiguous PV B-operand loads
  transpose_v<<<(NT * E) / 256, 256, 0, stream>>>(v, vt);

  // 4) flash attention: grid (T/16, B*H), one wave per block
  flash_attn<<<dim3(128, 32), 32, 0, stream>>>(q, k, vt, attn);

  // 5) O-projection fused with residual: xres = bf16(x + attn @ wo^T)
  gemm_bf16_nt<1><<<projBlocks, 256, 0, stream>>>(attn, wo, xres, xf, NT, E, E);

  // 6) logits = xres @ wfc^T -> f32 d_out (NT stores); blocks = 32*500 = 16000
  const int logitBlocks = (NT / 128) * (V / 64);
  gemm_bf16_nt<2><<<logitBlocks, 256, 0, stream>>>(xres, wfc, logits, nullptr, NT, V, E);
}
